// MultiHeadAttention_45062796870406
// MI455X (gfx1250) — matrix operand order repaired and checked by reading, not compiled, not run on hardware
//
#include <hip/hip_runtime.h>
#include <hip/hip_bf16.h>

// ---------------------------------------------------------------------------
// MI455X (gfx1250) multi-head attention with per-pair logit offsets.
// B=8, T=1024, D=512, H=8, K=V=64, DO=8, DM=512.
// All matmuls in bf16 WMMA (v_wmma_f32_16x16x32_bf16), f32 accumulate.
// 1/sqrt(K) is folded into the q projection; offset matvec + mask are fused
// into the cooperative logit_offset fill (read exactly once from HBM).
// ---------------------------------------------------------------------------

typedef __attribute__((ext_vector_type(16))) __bf16 v16bf;
typedef __attribute__((ext_vector_type(8)))  float  v8f;
typedef __attribute__((ext_vector_type(4)))  float  f32x4;
typedef __attribute__((ext_vector_type(2)))  float  f32x2;

union V16BF { v16bf v; f32x4 q[2]; };

// Build a 16-element bf16 A/B fragment from two 16-byte chunks.
__device__ inline v16bf load_bf16_2(const __bf16* p0, const __bf16* p1) {
    V16BF u;
    u.q[0] = *(const f32x4*)p0;
    u.q[1] = *(const f32x4*)p1;
    return u.v;
}

// f32 source -> bf16 A fragment (two 8-float chunks, K gap of 16).
__device__ inline v16bf cvt_f32_frag(const float* p0, const float* p1) {
    v16bf a;
#pragma unroll
    for (int i = 0; i < 8; ++i) {
        a[i]     = (__bf16)p0[i];
        a[i + 8] = (__bf16)p1[i];
    }
    return a;
}

__device__ inline v8f wmma_bf16(v16bf a, v16bf b, v8f c) {
    return __builtin_amdgcn_wmma_f32_16x16x32_bf16(
        /*neg_a=*/false, a, /*neg_b=*/false, b,
        /*c_mod=*/(short)0, c, /*reuse_a=*/false, /*reuse_b=*/false);
}

// Row reductions across the 16-lane halves of a wave32 (C-fragment layout:
// one matrix row lives in lanes 0..15 or 16..31 of one VGPR).
__device__ inline float red_max16(float x) {
    x = fmaxf(x, __shfl_xor(x, 1, 32));
    x = fmaxf(x, __shfl_xor(x, 2, 32));
    x = fmaxf(x, __shfl_xor(x, 4, 32));
    x = fmaxf(x, __shfl_xor(x, 8, 32));
    return x;
}
__device__ inline float red_sum16(float x) {
    x += __shfl_xor(x, 1, 32);
    x += __shfl_xor(x, 2, 32);
    x += __shfl_xor(x, 4, 32);
    x += __shfl_xor(x, 8, 32);
    return x;
}

// ---------------------------------------------------------------------------
// Kernel 0: transpose + convert the four 512x512 weight matrices to bf16
// column-major so WMMA B fragments are two contiguous 16B loads per lane.
// ---------------------------------------------------------------------------
__global__ __launch_bounds__(256) void prep_weights(
    const float* __restrict__ Wq, const float* __restrict__ Wk,
    const float* __restrict__ Wv, const float* __restrict__ Wout,
    __bf16* __restrict__ WqT, __bf16* __restrict__ WkT,
    __bf16* __restrict__ WvT, __bf16* __restrict__ WoutT) {
    int tid = blockIdx.x * 256 + threadIdx.x;     // 4 * 512 * 512 total
    int w = tid >> 18;
    int r = tid & ((1 << 18) - 1);
    int n = r >> 9;
    int k = r & 511;
    const float* W = (w == 0) ? Wq : (w == 1) ? Wk : (w == 2) ? Wv : Wout;
    __bf16* WT     = (w == 0) ? WqT : (w == 1) ? WkT : (w == 2) ? WvT : WoutT;
    WT[(size_t)n * 512 + k] = (__bf16)W[(size_t)k * 512 + n];
}

// ---------------------------------------------------------------------------
// WMMA GEMM:  C(8192x512) = A(8192x512) * B(512x512), BT is bf16 col-major.
// One wave computes a 16x64 tile (4 accumulators), K-loop step 32.
// MODE 0: store scale*C as bf16 to [B,H,T,64]  (q / k projection)
// MODE 1: store bf16 to [B,H,64,T]             (v projection, transposed)
// MODE 2: store f32 + bias to [M,N]            (final output GEMM, A bf16)
// ---------------------------------------------------------------------------
template <bool ABF16, int MODE>
__global__ __launch_bounds__(256) void gemm_wmma(
    const void* __restrict__ Aptr, const __bf16* __restrict__ BT,
    const float* __restrict__ bias, void* __restrict__ Cptr, float scale) {
    const int KD = 512, NN = 512;
    int wave = (blockIdx.x << 3) + (threadIdx.x >> 5);   // 0..4095
    int lane = threadIdx.x & 31;
    int tM = wave >> 3;         // 0..511 (M/16)
    int tN = wave & 7;          // 0..7   (N/64)
    int row = tM * 16 + (lane & 15);
    int abase = (lane < 16) ? 0 : 8;
    int kside = (lane < 16) ? 0 : 8;

    v8f acc[4];
#pragma unroll
    for (int g = 0; g < 4; ++g)
#pragma unroll
        for (int j = 0; j < 8; ++j) acc[g][j] = 0.f;

    for (int k0 = 0; k0 < KD; k0 += 32) {
        v16bf a;
        if (ABF16) {
            const __bf16* A = (const __bf16*)Aptr;
            const __bf16* pa = A + (size_t)row * KD + k0 + abase;
            a = load_bf16_2(pa, pa + 16);
        } else {
            const float* A = (const float*)Aptr;
            const float* pa = A + (size_t)row * KD + k0 + abase;
            a = cvt_f32_frag(pa, pa + 16);
        }
#pragma unroll
        for (int g = 0; g < 4; ++g) {
            int col = tN * 64 + g * 16 + (lane & 15);
            const __bf16* pb = BT + (size_t)col * KD + k0 + kside;
            v16bf b = load_bf16_2(pb, pb + 16);
            acc[g] = wmma_bf16(a, b, acc[g]);
        }
    }

    int mo = (lane < 16) ? 0 : 8;
#pragma unroll
    for (int g = 0; g < 4; ++g) {
        int col = tN * 64 + g * 16 + (lane & 15);
#pragma unroll
        for (int j = 0; j < 8; ++j) {
            int m = tM * 16 + j + mo;
            float v = acc[g][j];
            if (MODE == 2) {
                ((float*)Cptr)[(size_t)m * NN + col] = v + bias[col];
            } else {
                v *= scale;                        // 1/sqrt(K) for q, 1.0 else
                int b = m >> 10, t = m & 1023;     // T = 1024
                int h = col >> 6, d = col & 63;    // head size 64
                __bf16* C = (__bf16*)Cptr;
                if (MODE == 0)
                    C[(((size_t)(b * 8 + h) * 1024) + t) * 64 + d] = (__bf16)v;
                else
                    C[(((size_t)(b * 8 + h) * 64) + d) * 1024 + t] = (__bf16)v;
            }
        }
    }
}

// ---------------------------------------------------------------------------
// Flash attention. Workgroup = (batch b, 16-row query tile). 8 waves = 8
// heads. The cooperative fill phase streams the 16x32x8 logit_offset tile
// (the dominant HBM stream, read exactly once) and FUSES the per-head
// matvec (@Wo_off + bo_off) and the mask bias into it, so the per-wave hot
// loop does a single LDS read + add per logit.
// ---------------------------------------------------------------------------
__global__ __launch_bounds__(256) void attn_kernel(
    const __bf16* __restrict__ qbf, const __bf16* __restrict__ kbf,
    const __bf16* __restrict__ vT,
    const float* __restrict__ logit_offset, const unsigned char* __restrict__ mask,
    const float* __restrict__ Wo_off, const float* __restrict__ bo_off,
    __bf16* __restrict__ attn_out) {
    const int T = 1024, H = 8;
    __shared__ float  s_offh[8][16 * 32];   // finished offsets [h][m*32+n], 16 KB
    __shared__ __bf16 s_P[8][16 * 32];      // per-head P tile, 8 KB

    int tid  = threadIdx.x;
    int h    = tid >> 5;           // wave == head
    int lane = tid & 31;
    int b    = blockIdx.x >> 6;
    int q0   = (blockIdx.x & 63) << 4;

    // Uniform (scalar-resident) offset weights and biases for all heads.
    float wsc[8][8];               // wsc[c][h2]
    float bos[8];
#pragma unroll
    for (int c = 0; c < 8; ++c)
#pragma unroll
        for (int h2 = 0; h2 < 8; ++h2) wsc[c][h2] = Wo_off[c * H + h2];
#pragma unroll
    for (int h2 = 0; h2 < 8; ++h2) bos[h2] = bo_off[h2];

    int mr = lane & 15;
    int ab = (lane < 16) ? 0 : 8;     // A-fragment K base
    int mo = (lane < 16) ? 0 : 8;     // C-fragment row offset
    int nlo = lane & 15;

    // Persistent Q A-fragments (16 x 64, split into two 16x32 fragments).
    // q was pre-scaled by 1/sqrt(K) in the projection kernel.
    const __bf16* qp = qbf + (((size_t)(b * H + h) * T) + q0 + mr) * 64;
    v16bf qA0 = load_bf16_2(qp + ab,      qp + ab + 16);
    v16bf qA1 = load_bf16_2(qp + 32 + ab, qp + 32 + ab + 16);

    // Fill-phase indices: each thread owns (row fm, key positions p0,p0+1).
    int fm = tid >> 4;
    int p0 = (tid & 15) << 1;

    float mrow[8], lrow[8];
    v8f O[4];
#pragma unroll
    for (int j = 0; j < 8; ++j) { mrow[j] = -3.0e38f; lrow[j] = 0.f; }
#pragma unroll
    for (int g = 0; g < 4; ++g)
#pragma unroll
        for (int j = 0; j < 8; ++j) O[g][j] = 0.f;

    for (int kb = 0; kb < T; kb += 32) {
        // ---- cooperative fill: load offsets, fuse matvec + bias + mask ----
        {
            const float* src =
                logit_offset + (((size_t)(b * T + q0 + fm)) * T + kb + p0) * 8;
            union { f32x4 q[4]; float f[16]; } u;
            const f32x4* s4 = (const f32x4*)src;
            u.q[0] = s4[0]; u.q[1] = s4[1]; u.q[2] = s4[2]; u.q[3] = s4[3];
            __builtin_prefetch((const void*)(src + 256), 0, 0);  // next key block

            const unsigned char* mk =
                mask + ((size_t)(b * T + q0 + fm)) * T + kb + p0;
            float mb0 = mk[0] ? 0.f : -1.0e30f;
            float mb1 = mk[1] ? 0.f : -1.0e30f;

#pragma unroll
            for (int h2 = 0; h2 < 8; ++h2) {
                float o0 = bos[h2] + mb0;
                float o1 = bos[h2] + mb1;
#pragma unroll
                for (int c = 0; c < 8; ++c) {
                    o0 = fmaf(u.f[c],     wsc[c][h2], o0);
                    o1 = fmaf(u.f[c + 8], wsc[c][h2], o1);
                }
                f32x2 pr; pr[0] = o0; pr[1] = o1;
                *(f32x2*)&s_offh[h2][fm * 32 + p0] = pr;
            }
        }
        __syncthreads();

        // ---- S = Q * K^T for 32 keys (two 16x16 C fragments) ----
        v8f S[2];
        int kdim = (lane < 16) ? 0 : 8;
#pragma unroll
        for (int s = 0; s < 2; ++s) {
            const __bf16* kp =
                kbf + (((size_t)(b * H + h) * T) + kb + s * 16 + mr) * 64 + kdim;
            v16bf kB0 = load_bf16_2(kp,      kp + 16);   // dims 0..31
            v16bf kB1 = load_bf16_2(kp + 32, kp + 48);   // dims 32..63
            v8f z;
#pragma unroll
            for (int j = 0; j < 8; ++j) z[j] = 0.f;
            z    = wmma_bf16(qA0, kB0, z);
            S[s] = wmma_bf16(qA1, kB1, z);
        }

        // ---- logits = S + precomputed offset; online softmax; P -> LDS ----
#pragma unroll
        for (int j = 0; j < 8; ++j) {
            int m = j + mo;
            float lg0 = S[0][j] + s_offh[h][m * 32 + nlo];
            float lg1 = S[1][j] + s_offh[h][m * 32 + nlo + 16];
            float rm   = red_max16(fmaxf(lg0, lg1));
            float newm = fmaxf(mrow[j], rm);
            float alpha = __expf(mrow[j] - newm);
            mrow[j] = newm;
            float pv0 = __expf(lg0 - newm);
            float pv1 = __expf(lg1 - newm);
            // store P immediately (short live ranges; own-wave LDS ordering)
            s_P[h][m * 32 + nlo]      = (__bf16)pv0;
            s_P[h][m * 32 + nlo + 16] = (__bf16)pv1;
            float rs = red_sum16(pv0 + pv1);
            lrow[j] = lrow[j] * alpha + rs;
#pragma unroll
            for (int g = 0; g < 4; ++g) O[g][j] *= alpha;
        }
        __syncthreads();

        const __bf16* pp = &s_P[h][mr * 32 + ab];
        v16bf Pa = load_bf16_2(pp, pp + 16);

        // ---- O += P * V (V stored transposed: contiguous keys per dim) ----
        int tb = kb + ((lane < 16) ? 0 : 8);
#pragma unroll
        for (int g = 0; g < 4; ++g) {
            int d = g * 16 + nlo;
            const __bf16* vp = vT + ((size_t)(b * H + h) * 64 + d) * T + tb;
            v16bf vB = load_bf16_2(vp, vp + 16);
            O[g] = wmma_bf16(Pa, vB, O[g]);
        }
        __syncthreads();   // guard s_offh / s_P for next iteration
    }

    // ---- normalize and store bf16 attn output [B*T, H*V] ----
#pragma unroll
    for (int j = 0; j < 8; ++j) {
        int m = q0 + j + mo;
        float inv = 1.0f / lrow[j];
#pragma unroll
        for (int g = 0; g < 4; ++g) {
            int d = g * 16 + nlo;
            attn_out[((size_t)(b * T + m)) * 512 + h * 64 + d] =
                (__bf16)(O[g][j] * inv);
        }
    }
}

// ---------------------------------------------------------------------------
extern "C" void kernel_launch(void* const* d_in, const int* in_sizes, int n_in,
                              void* d_out, int out_size, void* d_ws, size_t ws_size,
                              hipStream_t stream) {
    (void)in_sizes; (void)n_in; (void)out_size; (void)ws_size;
    const float* query        = (const float*)d_in[0];
    const float* key_in       = (const float*)d_in[1];
    const float* value        = (const float*)d_in[2];
    const float* logit_offset = (const float*)d_in[3];
    const unsigned char* mask = (const unsigned char*)d_in[4];
    const float* Wq    = (const float*)d_in[5];
    const float* Wk    = (const float*)d_in[6];
    const float* Wv    = (const float*)d_in[7];
    const float* Wo_off= (const float*)d_in[8];
    const float* bo_off= (const float*)d_in[9];
    const float* Wout  = (const float*)d_in[10];
    const float* bout  = (const float*)d_in[11];

    const size_t szProj = (size_t)8 * 8 * 1024 * 64 * sizeof(__bf16);  // 8 MiB
    const size_t szAttn = (size_t)8 * 1024 * 512 * sizeof(__bf16);     // 8 MiB
    const size_t szW    = (size_t)512 * 512 * sizeof(__bf16);          // 512 KiB

    char* ws = (char*)d_ws;
    __bf16* qbf   = (__bf16*)ws;              ws += szProj;
    __bf16* kbf   = (__bf16*)ws;              ws += szProj;
    __bf16* vTbf  = (__bf16*)ws;              ws += szProj;
    __bf16* attn  = (__bf16*)ws;              ws += szAttn;
    __bf16* WqT   = (__bf16*)ws;              ws += szW;
    __bf16* WkT   = (__bf16*)ws;              ws += szW;
    __bf16* WvT   = (__bf16*)ws;              ws += szW;
    __bf16* WoutT = (__bf16*)ws;              ws += szW;

    // 0) weights -> bf16 column-major
    prep_weights<<<4096, 256, 0, stream>>>(Wq, Wk, Wv, Wout, WqT, WkT, WvT, WoutT);
    // 1-3) q/k/v projections (q pre-scaled by 1/sqrt(K); v transposed)
    gemm_wmma<false, 0><<<512, 256, 0, stream>>>(query,  WqT, nullptr, qbf, 0.125f);
    gemm_wmma<false, 0><<<512, 256, 0, stream>>>(key_in, WkT, nullptr, kbf, 1.0f);
    gemm_wmma<false, 1><<<512, 256, 0, stream>>>(value,  WvT, nullptr, vTbf, 1.0f);
    // 4) fused flash attention with precomputed logit offsets + mask
    attn_kernel<<<512, 256, 0, stream>>>(qbf, kbf, vTbf, logit_offset, mask,
                                         Wo_off, bo_off, attn);
    // 5) output projection with bias
    gemm_wmma<true, 2><<<512, 256, 0, stream>>>(attn, WoutT, bout,
                                                (float*)d_out, 1.0f);
}
